// TwentyConvPool_14242111553635
// MI455X (gfx1250) — matrix operand order repaired and checked by reading, not compiled, not run on hardware
//
#include <hip/hip_runtime.h>
#include <stddef.h>

// ---------------------------------------------------------------------------
// Types for CDNA5 WMMA (wave32): D = A(16x32 f16) * B(32x16 f16) + C(16x16 f32)
// ---------------------------------------------------------------------------
typedef _Float16 v16h __attribute__((ext_vector_type(16)));
typedef _Float16 v8h  __attribute__((ext_vector_type(8)));
typedef float    v8f  __attribute__((ext_vector_type(8)));

union HV { v16h v; v8h h[2]; _Float16 e[16]; };
union FV { v8f  v; float e[8]; };

#define ACT_NONE 0
#define ACT_RELU 1

struct SelState { unsigned prefix; int kRem; };

// ---------------------------------------------------------------------------
// Dense layer via WMMA with LDS staging:
//   Y[n,Nout] = act( X[n,Kin] @ W[Kin,Nout] + bias ),  Kin <= 64, Nout <= 64
// Block = 128 threads (4 waves), covers 64 rows. Whole W panel + X tile are
// staged f32->f16 into LDS with coalesced loads; fragments are then read as
// aligned 16B LDS vectors (ds_load_b128) per the ISA 7.12.2 layouts:
//   A: lane l, half j: M=l&15, K = (j<8 ? 8*(l>>4)+j : 16+8*(l>>4)+(j-8))
//      -> two contiguous 8-half runs at offsets 8*grp and 16+8*grp
//   B: lane l, half j: N=l&15, K as for A -> two contiguous 8-half runs at offsets 8*grp and 16+8*grp
//   C/D: lane l, vgpr r: M = r + 8*(l>>4), N = l&15
// ---------------------------------------------------------------------------
__global__ void wmma_linear_kernel(const float* __restrict__ X, int n, int Kin,
                                   const float* __restrict__ W, const float* __restrict__ bias,
                                   float* __restrict__ Y, int Nout, int act)
{
  __shared__ __align__(16) _Float16 ldsX[64 * 64];
  __shared__ __align__(16) _Float16 ldsW[64 * 64];

  const int nTilesN  = (Nout + 15) >> 4;
  const int nChunksK = (Kin  + 31) >> 5;
  const int Kpad = nChunksK * 32;          // 32 or 64
  const int Npad = nTilesN * 16;           // 16..64
  const int blockBase = blockIdx.x * 64;

  // ---- stage X tile (64 rows x Kpad) as f16, coalesced over k ----
  for (int t = threadIdx.x; t < 64 * Kpad; t += blockDim.x) {
    int r = t / Kpad, k = t - r * Kpad;
    int row = blockBase + r;
    ldsX[t] = (row < n && k < Kin) ? (_Float16)X[(size_t)row * Kin + k] : (_Float16)0.0f;
  }
  // ---- stage W panel (Kpad x Npad) as f16, K-contiguous per column ----
  for (int t = threadIdx.x; t < Kpad * Npad; t += blockDim.x) {
    int k = t / Npad, col = t - k * Npad;  // col fastest -> coalesced global read
    _Float16 v = (k < Kin && col < Nout) ? (_Float16)W[(size_t)k * Nout + col] : (_Float16)0.0f;
    ldsW[col * Kpad + k] = v;
  }
  __syncthreads();

  const int wave = threadIdx.x >> 5;
  const int lane = threadIdx.x & 31;
  const int grp  = lane >> 4;
  const int m    = lane & 15;
  const int tileBase = blockBase + wave * 16;
  if (tileBase >= n) return;               // after the barrier: safe

  for (int nt = 0; nt < nTilesN; ++nt) {
    FV c;
    #pragma unroll
    for (int r = 0; r < 8; ++r) c.e[r] = 0.0f;

    for (int kc = 0; kc < nChunksK; ++kc) {
      HV a, b;
      const _Float16* aBase = &ldsX[(wave * 16 + m) * Kpad + kc * 32];
      a.h[0] = *(const v8h*)(aBase + 8 * grp);        // K = 8*grp .. +7
      a.h[1] = *(const v8h*)(aBase + 16 + 8 * grp);   // K = 16+8*grp .. +7
      const _Float16* bBase = &ldsW[(nt * 16 + m) * Kpad + kc * 32];
      b.h[0] = *(const v8h*)(bBase + 8 * grp);        // K = 8*grp .. +7
      b.h[1] = *(const v8h*)(bBase + 16 + 8 * grp);   // K = 16+8*grp .. +7
      c.v = __builtin_amdgcn_wmma_f32_16x16x32_f16(false, a.v, false, b.v,
                                                   (short)0, c.v, false, false);
    }
    #pragma unroll
    for (int r = 0; r < 8; ++r) {
      int rr  = tileBase + 8 * grp + r;
      int col = nt * 16 + m;
      if (rr < n && col < Nout) {
        float v = c.e[r] + (bias ? bias[col] : 0.0f);
        if (act == ACT_RELU) v = fmaxf(v, 0.0f);
        Y[(size_t)rr * Nout + col] = v;
      }
    }
  }
}

// ---------------------------------------------------------------------------
// Edge setup: mask out existing self loops
// ---------------------------------------------------------------------------
__global__ void init_edges_kernel(const int* __restrict__ eidx, int* __restrict__ src,
                                  int* __restrict__ dst, float* __restrict__ mask, int E)
{
  int e = blockIdx.x * blockDim.x + threadIdx.x;
  if (e >= E) return;
  int s = eidx[e], d = eidx[E + e];
  src[e] = s; dst[e] = d;
  mask[e] = (s != d) ? 1.0f : 0.0f;
}

// ---------------------------------------------------------------------------
// FeaST edge scatter: per (edge, f): msg = sum_h softmax_h(ux[s]-ux[d]+c) * xw[s,h,f]
// ---------------------------------------------------------------------------
__global__ void feast_edge_kernel(const int* __restrict__ src, const int* __restrict__ dst,
                                  const float* __restrict__ mask, const float* __restrict__ ux,
                                  const float* __restrict__ cpar, const float* __restrict__ xw,
                                  float* __restrict__ agg, float* __restrict__ cnt,
                                  int E, int F, int fshift)
{
  long long t = (long long)blockIdx.x * blockDim.x + threadIdx.x;
  if (t >= ((long long)E << fshift)) return;
  int e = (int)(t >> fshift);
  int f = (int)(t & (F - 1));
  if (e + 4096 < E) {                 // stream-ahead: gfx1250 global_prefetch_b8
    __builtin_prefetch(&src[e + 4096], 0, 1);
    __builtin_prefetch(&dst[e + 4096], 0, 1);
    __builtin_prefetch(&mask[e + 4096], 0, 1);
  }
  float mk = mask[e];
  if (mk == 0.0f) return;
  int s = src[e], d = dst[e];
  float z0 = ux[s * 4 + 0] - ux[d * 4 + 0] + cpar[0];
  float z1 = ux[s * 4 + 1] - ux[d * 4 + 1] + cpar[1];
  float z2 = ux[s * 4 + 2] - ux[d * 4 + 2] + cpar[2];
  float z3 = ux[s * 4 + 3] - ux[d * 4 + 3] + cpar[3];
  float mx = fmaxf(fmaxf(z0, z1), fmaxf(z2, z3));
  float q0 = __expf(z0 - mx), q1 = __expf(z1 - mx), q2 = __expf(z2 - mx), q3 = __expf(z3 - mx);
  float inv = 1.0f / (q0 + q1 + q2 + q3);
  const float* xws = xw + ((size_t)s * 4) * F + f;
  float msg = (q0 * xws[0] + q1 * xws[F] + q2 * xws[2 * F] + q3 * xws[3 * F]) * inv;
  atomicAdd(&agg[(size_t)d * F + f], mk * msg);
  if (f == 0) atomicAdd(&cnt[d], mk);
}

// ---------------------------------------------------------------------------
// FeaST finalize: add analytic self-loop message (q = softmax(c)), mean, bias, relu
// ---------------------------------------------------------------------------
__global__ void feast_finalize_kernel(const float* __restrict__ agg, const float* __restrict__ cnt,
                                      const float* __restrict__ xw, const float* __restrict__ cpar,
                                      const float* __restrict__ bias, float* __restrict__ out,
                                      int n, int F, int fshift, int relu)
{
  int t = blockIdx.x * blockDim.x + threadIdx.x;
  if (t >= (n << fshift)) return;
  int i = t >> fshift;
  int f = t & (F - 1);
  float c0 = cpar[0], c1 = cpar[1], c2 = cpar[2], c3 = cpar[3];
  float mx = fmaxf(fmaxf(c0, c1), fmaxf(c2, c3));
  float q0 = __expf(c0 - mx), q1 = __expf(c1 - mx), q2 = __expf(c2 - mx), q3 = __expf(c3 - mx);
  float inv = 1.0f / (q0 + q1 + q2 + q3);
  const float* xwi = xw + ((size_t)i * 4) * F + f;
  float self = (q0 * xwi[0] + q1 * xwi[F] + q2 * xwi[2 * F] + q3 * xwi[3 * F]) * inv;
  float denom = cnt[i] + 1.0f;                      // >= 1, matches max(cnt,1)
  float v = (agg[(size_t)i * F + f] + self) / denom + bias[f];
  if (relu) v = fmaxf(v, 0.0f);
  out[(size_t)i * F + f] = v;
}

// ---------------------------------------------------------------------------
// BatchNorm over nodes, C==4 always in this network
// ---------------------------------------------------------------------------
__global__ void bn_reduce4_kernel(const float* __restrict__ x, int n, float* __restrict__ stats)
{
  __shared__ float ss[8];
  if (threadIdx.x < 8) ss[threadIdx.x] = 0.0f;
  __syncthreads();
  float s0 = 0, s1 = 0, s2 = 0, s3 = 0, p0 = 0, p1 = 0, p2 = 0, p3 = 0;
  for (int i = blockIdx.x * blockDim.x + threadIdx.x; i < n; i += gridDim.x * blockDim.x) {
    float v0 = x[i * 4 + 0], v1 = x[i * 4 + 1], v2 = x[i * 4 + 2], v3 = x[i * 4 + 3];
    s0 += v0; s1 += v1; s2 += v2; s3 += v3;
    p0 += v0 * v0; p1 += v1 * v1; p2 += v2 * v2; p3 += v3 * v3;
  }
  atomicAdd(&ss[0], s0); atomicAdd(&ss[1], s1); atomicAdd(&ss[2], s2); atomicAdd(&ss[3], s3);
  atomicAdd(&ss[4], p0); atomicAdd(&ss[5], p1); atomicAdd(&ss[6], p2); atomicAdd(&ss[7], p3);
  __syncthreads();
  if (threadIdx.x < 8) atomicAdd(&stats[threadIdx.x], ss[threadIdx.x]);
}

__global__ void bn_apply4_kernel(float* __restrict__ x, int n, const float* __restrict__ stats,
                                 const float* __restrict__ g, const float* __restrict__ b)
{
  int i = blockIdx.x * blockDim.x + threadIdx.x;
  if (i >= n) return;
  float rn = 1.0f / (float)n;
  #pragma unroll
  for (int ch = 0; ch < 4; ++ch) {
    float mu  = stats[ch] * rn;
    float var = stats[4 + ch] * rn - mu * mu;
    x[i * 4 + ch] = (x[i * 4 + ch] - mu) * rsqrtf(var + 1e-5f) * g[ch] + b[ch];
  }
}

// ---------------------------------------------------------------------------
// TopK pooling (ratio .5): score = tanh(x@w/||w||), radix-select kth largest
// ---------------------------------------------------------------------------
__global__ void wnorm_kernel(const float* __restrict__ w, float* __restrict__ invn)
{
  if (threadIdx.x == 0 && blockIdx.x == 0) {
    float a = 0.0f;
    #pragma unroll
    for (int f = 0; f < 16; ++f) a += w[f] * w[f];
    invn[0] = rsqrtf(a);
  }
}

__global__ void pool_score_kernel(const float* __restrict__ x, const float* __restrict__ w,
                                  const float* __restrict__ invn, float* __restrict__ score,
                                  unsigned* __restrict__ keys, int n)
{
  int i = blockIdx.x * blockDim.x + threadIdx.x;
  if (i >= n) return;
  float acc = 0.0f;
  #pragma unroll
  for (int f = 0; f < 16; ++f) acc += x[(size_t)i * 16 + f] * w[f];
  float s = tanhf(acc * invn[0]);
  score[i] = s;
  unsigned b = __float_as_uint(s);
  keys[i] = (b & 0x80000000u) ? ~b : (b | 0x80000000u);   // order-preserving key
}

__global__ void sel_init_kernel(SelState* sel, int k)
{
  if (threadIdx.x == 0 && blockIdx.x == 0) { sel->prefix = 0u; sel->kRem = k; }
}

__global__ void sel_hist_kernel(const unsigned* __restrict__ keys, int n,
                                const SelState* __restrict__ sel, int shift, int* __restrict__ hist)
{
  int i = blockIdx.x * blockDim.x + threadIdx.x;
  if (i >= n) return;
  unsigned hm = (shift + 8 >= 32) ? 0u : (~0u << (shift + 8));
  unsigned u = keys[i];
  if ((u & hm) == (sel->prefix & hm)) atomicAdd(&hist[(u >> shift) & 255u], 1);
}

__global__ void sel_pick_kernel(SelState* sel, const int* __restrict__ hist, int shift)
{
  if (threadIdx.x == 0 && blockIdx.x == 0) {
    int kr = sel->kRem, cum = 0;
    for (int d = 255; d >= 0; --d) {
      int c = hist[d];
      if (cum + c >= kr) { sel->prefix |= ((unsigned)d) << shift; sel->kRem = kr - cum; break; }
      cum += c;
    }
  }
}

__global__ void pool_flag_kernel(const unsigned* __restrict__ keys, int n,
                                 const SelState* __restrict__ sel,
                                 int* __restrict__ gflag, int* __restrict__ eflag)
{
  int i = blockIdx.x * blockDim.x + threadIdx.x;
  if (i >= n) return;
  unsigned T = sel->prefix, u = keys[i];
  gflag[i] = (u > T) ? 1 : 0;
  eflag[i] = (u == T) ? 1 : 0;
}

// exclusive scan: block pass, serial block-sum scan, add pass (deterministic)
__global__ void scan_block_kernel(const int* __restrict__ in, int* __restrict__ out,
                                  int* __restrict__ bsum, int n)
{
  __shared__ int sh[256];
  int i = blockIdx.x * 256 + threadIdx.x;
  int v = (i < n) ? in[i] : 0;
  sh[threadIdx.x] = v;
  __syncthreads();
  for (int off = 1; off < 256; off <<= 1) {
    int t = (threadIdx.x >= off) ? sh[threadIdx.x - off] : 0;
    __syncthreads();
    sh[threadIdx.x] += t;
    __syncthreads();
  }
  if (i < n) out[i] = sh[threadIdx.x] - v;
  if (threadIdx.x == 255) bsum[blockIdx.x] = sh[255];
}

__global__ void scan_bsum_kernel(int* bsum, int nb)
{
  if (threadIdx.x == 0 && blockIdx.x == 0) {
    int run = 0;
    for (int b = 0; b < nb; ++b) { int t = bsum[b]; bsum[b] = run; run += t; }
  }
}

__global__ void scan_add_kernel(int* __restrict__ out, const int* __restrict__ bsum, int n)
{
  int i = blockIdx.x * 256 + threadIdx.x;
  if (i < n) out[i] += bsum[blockIdx.x];
}

__global__ void pool_compact_kernel(const float* __restrict__ xin, const float* __restrict__ score,
                                    const unsigned* __restrict__ keys, const SelState* __restrict__ sel,
                                    const int* __restrict__ gscan, const int* __restrict__ escan,
                                    int* __restrict__ newidx, float* __restrict__ xout, int n, int k)
{
  int i = blockIdx.x * blockDim.x + threadIdx.x;
  if (i >= n) return;
  unsigned T = sel->prefix;
  int kRem = sel->kRem;
  int nG = k - kRem;
  unsigned u = keys[i];
  int nw = -1;
  if (u > T)                               nw = gscan[i];
  else if (u == T && escan[i] < kRem)      nw = nG + escan[i];
  newidx[i] = nw;
  if (nw >= 0) {
    float s = score[i];
    #pragma unroll
    for (int f = 0; f < 16; ++f) xout[(size_t)nw * 16 + f] = xin[(size_t)i * 16 + f] * s;
  }
}

__global__ void remap_edges_kernel(int* __restrict__ src, int* __restrict__ dst,
                                   float* __restrict__ mask, const int* __restrict__ newidx, int E)
{
  int e = blockIdx.x * blockDim.x + threadIdx.x;
  if (e >= E) return;
  int s = newidx[src[e]], d = newidx[dst[e]];
  float m = mask[e];
  if (s < 0 || d < 0) m = 0.0f;
  src[e] = (s < 0) ? 0 : s;
  dst[e] = (d < 0) ? 0 : d;
  mask[e] = m;
}

// ---------------------------------------------------------------------------
// Final 16 -> 1 linear + sigmoid
// ---------------------------------------------------------------------------
__global__ void final_out_kernel(const float* __restrict__ z, const float* __restrict__ w,
                                 const float* __restrict__ b, float* __restrict__ out, int n)
{
  int i = blockIdx.x * blockDim.x + threadIdx.x;
  if (i >= n) return;
  float a = 0.0f;
  #pragma unroll
  for (int f = 0; f < 16; ++f) a += z[(size_t)i * 16 + f] * w[f];
  a += b[0];
  out[i] = 1.0f / (1.0f + __expf(-a));
}

// ---------------------------------------------------------------------------
// Host orchestration
// ---------------------------------------------------------------------------
struct FeastP { const float *W, *u, *c, *b; };

struct WSP {
  float *xA, *xB, *xw, *ux, *agg, *cnt;
  int *src, *dst; float *mask;
  float *score; unsigned *keys;
  int *gflag, *eflag, *gscan, *escan, *bsum, *newidx;
  float *stats, *invn; SelState *sel; int *hist;
  float *z1, *z2, *z3;
  int E;
};

static void feast(hipStream_t st, WSP& w, const float* xin, int n, int cin, int F,
                  const FeastP& p, float* xout, int relu)
{
  int blocks = (n + 63) / 64;
  wmma_linear_kernel<<<blocks, 128, 0, st>>>(xin, n, cin, p.W, nullptr, w.xw, 4 * F, ACT_NONE);
  wmma_linear_kernel<<<blocks, 128, 0, st>>>(xin, n, cin, p.u, nullptr, w.ux, 4, ACT_NONE);
  hipMemsetAsync(w.agg, 0, (size_t)n * F * sizeof(float), st);
  hipMemsetAsync(w.cnt, 0, (size_t)n * sizeof(float), st);
  int fshift = (F == 16) ? 4 : 2;
  long long tot = (long long)w.E << fshift;
  feast_edge_kernel<<<(int)((tot + 255) / 256), 256, 0, st>>>(
      w.src, w.dst, w.mask, w.ux, p.c, w.xw, w.agg, w.cnt, w.E, F, fshift);
  feast_finalize_kernel<<<((n << fshift) + 255) / 256, 256, 0, st>>>(
      w.agg, w.cnt, w.xw, p.c, p.b, xout, n, F, fshift, relu);
}

static void scan_exclusive(const int* in, int* out, int* bsum, int n, hipStream_t st)
{
  int nb = (n + 255) / 256;
  scan_block_kernel<<<nb, 256, 0, st>>>(in, out, bsum, n);
  scan_bsum_kernel<<<1, 32, 0, st>>>(bsum, nb);
  scan_add_kernel<<<nb, 256, 0, st>>>(out, bsum, n);
}

static void pool(hipStream_t st, WSP& w, const float* xin, int n, int k,
                 const float* pw, float* xout)
{
  int nb = (n + 255) / 256;
  wnorm_kernel<<<1, 32, 0, st>>>(pw, w.invn);
  pool_score_kernel<<<nb, 256, 0, st>>>(xin, pw, w.invn, w.score, w.keys, n);
  sel_init_kernel<<<1, 32, 0, st>>>(w.sel, k);
  for (int p = 0; p < 4; ++p) {
    int shift = 24 - 8 * p;
    hipMemsetAsync(w.hist, 0, 256 * sizeof(int), st);
    sel_hist_kernel<<<nb, 256, 0, st>>>(w.keys, n, w.sel, shift, w.hist);
    sel_pick_kernel<<<1, 32, 0, st>>>(w.sel, w.hist, shift);
  }
  pool_flag_kernel<<<nb, 256, 0, st>>>(w.keys, n, w.sel, w.gflag, w.eflag);
  scan_exclusive(w.gflag, w.gscan, w.bsum, n, st);
  scan_exclusive(w.eflag, w.escan, w.bsum, n, st);
  pool_compact_kernel<<<nb, 256, 0, st>>>(xin, w.score, w.keys, w.sel,
                                          w.gscan, w.escan, w.newidx, xout, n, k);
  remap_edges_kernel<<<(w.E + 255) / 256, 256, 0, st>>>(w.src, w.dst, w.mask, w.newidx, w.E);
}

static void bnorm(hipStream_t st, WSP& w, float* x, int n, const float* g, const float* b)
{
  hipMemsetAsync(w.stats, 0, 8 * sizeof(float), st);
  bn_reduce4_kernel<<<256, 256, 0, st>>>(x, n, w.stats);
  bn_apply4_kernel<<<(n + 255) / 256, 256, 0, st>>>(x, n, w.stats, g, b);
}

extern "C" void kernel_launch(void* const* d_in, const int* in_sizes, int n_in,
                              void* d_out, int out_size, void* d_ws, size_t ws_size,
                              hipStream_t stream)
{
  const float* x_in = (const float*)d_in[0];
  const int*   eidx = (const int*)d_in[1];
  int n0 = in_sizes[0] / 16;
  int E  = in_sizes[1] / 2;

  // ---- parse params (recursive insertion order of the setup dict) ----
  int idx = 2;
  auto nf = [&]() { return (const float*)d_in[idx++]; };
  FeastP B[5][4];
  const float *gpar[5], *bpar[5], *pw[3];
  for (int blk = 0; blk < 5; ++blk) {
    for (int c = 0; c < 4; ++c) { B[blk][c].W = nf(); B[blk][c].u = nf(); B[blk][c].c = nf(); B[blk][c].b = nf(); }
    gpar[blk] = nf(); bpar[blk] = nf();
    if (blk < 3) pw[blk] = nf();
  }
  const float* l1w = nf(); const float* l1b = nf();
  const float* l2w = nf(); const float* l2b = nf();
  const float* l3w = nf(); const float* l3b = nf();
  const float* low = nf(); const float* lob = nf();

  // ---- carve workspace ----
  char* p = (char*)d_ws;
  auto alloc = [&](size_t bytes) -> void* { void* r = (void*)p; p += (bytes + 255) & ~(size_t)255; return r; };
  WSP w;
  w.E      = E;
  w.xA     = (float*)alloc((size_t)n0 * 16 * 4);
  w.xB     = (float*)alloc((size_t)n0 * 16 * 4);
  w.xw     = (float*)alloc((size_t)n0 * 64 * 4);
  w.ux     = (float*)alloc((size_t)n0 * 4 * 4);
  w.agg    = (float*)alloc((size_t)n0 * 16 * 4);
  w.cnt    = (float*)alloc((size_t)n0 * 4);
  w.src    = (int*)alloc((size_t)E * 4);
  w.dst    = (int*)alloc((size_t)E * 4);
  w.mask   = (float*)alloc((size_t)E * 4);
  w.score  = (float*)alloc((size_t)n0 * 4);
  w.keys   = (unsigned*)alloc((size_t)n0 * 4);
  w.gflag  = (int*)alloc((size_t)n0 * 4);
  w.eflag  = (int*)alloc((size_t)n0 * 4);
  w.gscan  = (int*)alloc((size_t)n0 * 4);
  w.escan  = (int*)alloc((size_t)n0 * 4);
  w.bsum   = (int*)alloc(2048 * 4);
  w.newidx = (int*)alloc((size_t)n0 * 4);
  w.stats  = (float*)alloc(64 * 4);
  w.invn   = (float*)alloc(64);
  w.sel    = (SelState*)alloc(64);
  w.hist   = (int*)alloc(256 * 4);
  int n3max = n0 / 8 + 8;
  w.z1 = (float*)alloc((size_t)n3max * 64 * 4);
  w.z2 = (float*)alloc((size_t)n3max * 64 * 4);
  w.z3 = (float*)alloc((size_t)n3max * 16 * 4);
  (void)ws_size; (void)n_in;

  init_edges_kernel<<<(E + 255) / 256, 256, 0, stream>>>(eidx, w.src, w.dst, w.mask, E);

  int n = n0;
  const float* cur = x_in;

  // ---- blocks 1..3: conv-pool (cin = 16 then 4; mid=16; cout=4) ----
  int cin = 16;
  for (int blk = 0; blk < 3; ++blk) {
    feast(stream, w, cur, n, cin, 16, B[blk][0], w.xA, 1); cur = w.xA;
    feast(stream, w, cur, n, 16,  16, B[blk][1], w.xB, 1); cur = w.xB;
    feast(stream, w, cur, n, 16,  16, B[blk][2], w.xA, 0); cur = w.xA;   // no relu before pool
    int k = (n + 1) / 2;
    pool(stream, w, cur, n, k, pw[blk], w.xB); cur = w.xB; n = k;
    feast(stream, w, cur, n, 16, 4, B[blk][3], w.xA, 1); cur = w.xA;
    bnorm(stream, w, w.xA, n, gpar[blk], bpar[blk]);
    cin = 4;
  }

  // ---- blocks 4..5: conv only (mid=4, cout=4), relu on all four ----
  for (int blk = 3; blk < 5; ++blk) {
    feast(stream, w, cur, n, 4, 4, B[blk][0], w.xB, 1); cur = w.xB;
    feast(stream, w, cur, n, 4, 4, B[blk][1], w.xA, 1); cur = w.xA;
    feast(stream, w, cur, n, 4, 4, B[blk][2], w.xB, 1); cur = w.xB;
    feast(stream, w, cur, n, 4, 4, B[blk][3], w.xA, 1); cur = w.xA;
    bnorm(stream, w, w.xA, n, gpar[blk], bpar[blk]);
  }

  // ---- MLP head (WMMA) + sigmoid ----
  int blocks = (n + 63) / 64;
  wmma_linear_kernel<<<blocks, 128, 0, stream>>>(cur,  n, 4,  l1w, l1b, w.z1, 64, ACT_RELU);
  wmma_linear_kernel<<<blocks, 128, 0, stream>>>(w.z1, n, 64, l2w, l2b, w.z2, 64, ACT_RELU);
  wmma_linear_kernel<<<blocks, 128, 0, stream>>>(w.z2, n, 64, l3w, l3b, w.z3, 16, ACT_RELU);
  final_out_kernel<<<(n + 255) / 256, 256, 0, stream>>>(w.z3, low, lob, (float*)d_out, n);
  (void)out_size;
}
